// LlamaStyleAttention_36962488549531
// MI455X (gfx1250) — matrix operand order repaired and checked by reading, not compiled, not run on hardware
//
#include <hip/hip_runtime.h>

#define SEQ 2048
#define DM 4096
#define NH 32
#define NKV 8
#define HD 128
#define KVD 1024  // NKV*HD

typedef __attribute__((ext_vector_type(16))) __bf16 v16bf;
typedef __attribute__((ext_vector_type(8)))  __bf16 v8bf;
typedef __attribute__((ext_vector_type(8)))  float  v8f;

// ---- WMMA fragment helpers (wave32; A/B use contiguous k = 16h + e, not the ISA table order) ----
// A-frag (16x32 bf16, M x K): lane L (m=L&15, h=L>>4) holds
//   elems 0..15 = A[m][base + 16h + e]
__device__ __forceinline__ v16bf load_a_frag(const __bf16* p) {
  union { v16bf v; v8bf h[2]; } u;
  u.h[0] = *(const v8bf*)(p);
  u.h[1] = *(const v8bf*)(p + 8);
  return u.v;
}
// B-frag (32x16 bf16, K x N): lane L (n=L&15, h=L>>4) holds
//   elems 0..15 = B[base + 16h + e][n]  -> for B = W^T this is W[n][base+16h+e]
__device__ __forceinline__ v16bf load_b_frag(const __bf16* p) {
  return *(const v16bf*)(p);
}
__device__ __forceinline__ v8f wmma_bf16(v16bf a, v16bf b, v8f c) {
  return __builtin_amdgcn_wmma_f32_16x16x32_bf16(false, a, false, b, (short)0, c,
                                                 false, false);
}

// ---- f32 -> bf16 elementwise convert ----
__global__ void f32_to_bf16(const float* __restrict__ in, __bf16* __restrict__ out,
                            size_t n) {
  size_t i = (size_t)blockIdx.x * blockDim.x + threadIdx.x;
  size_t stride = (size_t)gridDim.x * blockDim.x;
  for (; i < n; i += stride) out[i] = (__bf16)in[i];
}

// ---- C[M][N] f32 = A[M][K] bf16 @ W[N][K]^T bf16 ----
// block = 256 threads = 8 waves; wave computes a 32(M) x 64(N) tile
// (two A-frags share four B-frags -> 8 WMMA per 32-K step, 24 B/lane/WMMA).
// block tile = 256(M) x 64(N); grid = (N/64, M/256)
__global__ __launch_bounds__(256) void gemm_xwT(const __bf16* __restrict__ A,
                                                const __bf16* __restrict__ W,
                                                float* __restrict__ C,
                                                int M, int N, int K) {
  const int lane = threadIdx.x & 31;
  const int wave = threadIdx.x >> 5;
  const int h = lane >> 4;
  const int r = lane & 15;
  const int m0 = blockIdx.y * 256 + wave * 32;
  const int n0 = blockIdx.x * 64;

  const __bf16* ap0 = A + (size_t)(m0 + r) * K + 16 * h;
  const __bf16* ap1 = A + (size_t)(m0 + 16 + r) * K + 16 * h;
  const __bf16* wp  = W + (size_t)(n0 + r) * K + 16 * h;

  v8f acc[2][4];
#pragma unroll
  for (int u = 0; u < 2; ++u)
#pragma unroll
    for (int t = 0; t < 4; ++t)
#pragma unroll
      for (int j = 0; j < 8; ++j) acc[u][t][j] = 0.0f;

  for (int kk = 0; kk < K; kk += 32) {
    __builtin_prefetch(ap0 + kk + 512, 0, 1);  // gfx1250 global_prefetch_b8
    __builtin_prefetch(ap1 + kk + 512, 0, 1);
    v16bf a0 = load_a_frag(ap0 + kk);
    v16bf a1 = load_a_frag(ap1 + kk);
#pragma unroll
    for (int t = 0; t < 4; ++t) {
      v16bf b = load_b_frag(wp + (size_t)t * 16 * K + kk);
      acc[0][t] = wmma_bf16(a0, b, acc[0][t]);
      acc[1][t] = wmma_bf16(a1, b, acc[1][t]);
    }
  }
  // C/D layout: vgpr j -> row m0 + u*16 + j + 8h, col n0 + t*16 + r
#pragma unroll
  for (int u = 0; u < 2; ++u) {
#pragma unroll
    for (int t = 0; t < 4; ++t) {
      float* cp = C + (size_t)(m0 + u * 16 + 8 * h) * N + n0 + t * 16 + r;
#pragma unroll
      for (int j = 0; j < 8; ++j) cp[(size_t)j * N] = acc[u][t][j];
    }
  }
}

// ---- RoPE + pack [S][nh*128] f32 -> [nh][S][128] bf16 ----
__global__ void rope_pack(const float* __restrict__ src, __bf16* __restrict__ dst,
                          int nheads) {
  const int s = blockIdx.x;
  const int hd = blockIdx.y;
  const int i = threadIdx.x;  // 0..63, rotates pair (2i, 2i+1)
  const size_t base = (size_t)s * ((size_t)nheads * HD) + (size_t)hd * HD;
  const float e = src[base + 2 * i];
  const float o = src[base + 2 * i + 1];
  const float inv = __powf(10000.0f, -(float)i * (1.0f / 64.0f));
  float sn, cs;
  __sincosf((float)s * inv, &sn, &cs);
  __bf16* d = dst + ((size_t)hd * SEQ + s) * HD;
  d[2 * i] = (__bf16)(e * cs - o * sn);
  d[2 * i + 1] = (__bf16)(e * sn + o * cs);
}

// ---- V pack+transpose: [S][KVD] f32 -> vt[kvh][d][S] bf16 ----
__global__ void pack_vt(const float* __restrict__ v, __bf16* __restrict__ vtb) {
  const int s = blockIdx.x;
  const int kvh = blockIdx.y;
  const int d = threadIdx.x;  // 0..127
  vtb[((size_t)kvh * HD + d) * SEQ + s] =
      (__bf16)v[(size_t)s * KVD + (size_t)kvh * HD + d];
}

// ---- Flash attention (causal, GQA 4:1), transposed-score formulation ----
// grid = (SEQ/128, NH), block = 256 (8 waves); each wave owns 16 queries.
// Scores computed transposed: T = K_block @ Q^T  (lane owns a query column),
// output accumulated transposed: O^T = V^T @ P^T (lane-uniform 1/l epilogue).
__global__ __launch_bounds__(256) void attn(const __bf16* __restrict__ qh,
                                            const __bf16* __restrict__ kh,
                                            const __bf16* __restrict__ vt,
                                            __bf16* __restrict__ ob) {
  const int lane = threadIdx.x & 31;
  const int wave = threadIdx.x >> 5;
  const int h = lane >> 4;
  const int r = lane & 15;
  const int head = blockIdx.y;
  const int kvh = head >> 2;  // n_rep = 4
  const int q0 = blockIdx.x * 128 + wave * 16;
  const int qrow = q0 + r;

  // Q fragments (B operand of T = K @ Q^T), 4 chunks of K-dim 32
  v16bf qf[4];
  const __bf16* qp = qh + ((size_t)head * SEQ + q0 + r) * HD + 16 * h;
#pragma unroll
  for (int kc = 0; kc < 4; ++kc) qf[kc] = load_b_frag(qp + kc * 32);

  v8f o[8];
#pragma unroll
  for (int dt = 0; dt < 8; ++dt)
#pragma unroll
    for (int j = 0; j < 8; ++j) o[dt][j] = 0.0f;

  float m_i = -1e30f, l_i = 0.0f;
  const float scale = 0.08838834764831845f;  // 1/sqrt(128)

  const __bf16* kp = kh + (size_t)kvh * SEQ * HD + 16 * h;
  const __bf16* vp = vt + (size_t)kvh * HD * SEQ + 16 * h;

  for (int kb = 0; kb < q0 + 16; kb += 32) {
    v8f t0, t1;
#pragma unroll
    for (int j = 0; j < 8; ++j) { t0[j] = 0.0f; t1[j] = 0.0f; }
#pragma unroll
    for (int kc = 0; kc < 4; ++kc) {
      v16bf k0 = load_a_frag(kp + (size_t)(kb + r) * HD + kc * 32);
      v16bf k1 = load_a_frag(kp + (size_t)(kb + 16 + r) * HD + kc * 32);
      t0 = wmma_bf16(k0, qf[kc], t0);
      t1 = wmma_bf16(k1, qf[kc], t1);
    }
    // t0[j] = score[key kb+j+8h][query qrow]; t1: keys +16. Mask + scale.
    float s0[8], s1[8], mx = -1e30f;
#pragma unroll
    for (int j = 0; j < 8; ++j) {
      const int key0 = kb + 8 * h + j;
      s0[j] = (key0 <= qrow) ? t0[j] * scale : -1e30f;
      s1[j] = (key0 + 16 <= qrow) ? t1[j] * scale : -1e30f;
      mx = fmaxf(mx, fmaxf(s0[j], s1[j]));
    }
    mx = fmaxf(mx, __shfl_xor(mx, 16, 32));  // combine key halves per query
    const float m_new = fmaxf(m_i, mx);
    const float alpha = __expf(m_i - m_new);
    float p0[8], p1[8], rsum = 0.0f;
#pragma unroll
    for (int j = 0; j < 8; ++j) {
      p0[j] = __expf(s0[j] - m_new);
      p1[j] = __expf(s1[j] - m_new);
      rsum += p0[j] + p1[j];
    }
    rsum += __shfl_xor(rsum, 16, 32);
    l_i = l_i * alpha + rsum;
    m_i = m_new;

    // Build P^T B-frag: lane(n,h) needs keys 16h..16h+15 of query n.
    // Local halves cover keys {8h..8h+7} (t0) and {16+8h..} (t1); fetch the
    // partner lane's half with shfl_xor(16).
    union { v16bf v; __bf16 e[16]; } pf;
#pragma unroll
    for (int j = 0; j < 8; ++j) {
      const float q0v = __shfl_xor(p0[j], 16, 32);  // partner's t0 half
      const float q1v = __shfl_xor(p1[j], 16, 32);  // partner's t1 half
      pf.e[j]     = (__bf16)(h ? q1v : p0[j]);   // keys 16h + j
      pf.e[j + 8] = (__bf16)(h ? p1[j] : q0v);   // keys 16h + 8 + j
    }
    // O^T += V^T(16d x 32k) @ P^T(32k x 16q) per d-tile
#pragma unroll
    for (int dt = 0; dt < 8; ++dt) {
      v16bf vf = load_a_frag(vp + (size_t)(dt * 16 + r) * SEQ + kb);
#pragma unroll
      for (int j = 0; j < 8; ++j) o[dt][j] *= alpha;
      o[dt] = wmma_bf16(vf, pf.v, o[dt]);
    }
  }

  // o[dt][j] = O^T[d = dt*16 + j + 8h][q = qrow]; lane-uniform normalize,
  // contiguous 16B bf16 stores into ob[q][head*128 + d].
  const float inv_l = 1.0f / l_i;
  __bf16* op = ob + (size_t)qrow * DM + head * HD + 8 * h;
#pragma unroll
  for (int dt = 0; dt < 8; ++dt) {
    union { v8bf v; __bf16 e[8]; } w;
#pragma unroll
    for (int j = 0; j < 8; ++j) w.e[j] = (__bf16)(o[dt][j] * inv_l);
    *(v8bf*)(op + dt * 16) = w.v;
  }
}

extern "C" void kernel_launch(void* const* d_in, const int* in_sizes, int n_in,
                              void* d_out, int out_size, void* d_ws, size_t ws_size,
                              hipStream_t stream) {
  const float* x  = (const float*)d_in[0];
  const float* wq = (const float*)d_in[1];
  const float* wk = (const float*)d_in[2];
  const float* wv = (const float*)d_in[3];
  const float* wo = (const float*)d_in[4];
  // d_in[5] attn_mask is tril(ones) -> causal mask computed analytically.

  char* ws = (char*)d_ws;
  size_t off = 0;
  auto alloc = [&](size_t bytes) -> char* {
    char* p = ws + off;
    off += (bytes + 255) & ~(size_t)255;
    return p;
  };
  __bf16* xb  = (__bf16*)alloc((size_t)SEQ * DM * 2);
  __bf16* wqb = (__bf16*)alloc((size_t)DM * DM * 2);
  __bf16* wkb = (__bf16*)alloc((size_t)KVD * DM * 2);
  __bf16* wvb = (__bf16*)alloc((size_t)KVD * DM * 2);
  __bf16* wob = (__bf16*)alloc((size_t)DM * DM * 2);
  float* qf32 = (float*)alloc((size_t)SEQ * DM * 4);
  float* kf32 = (float*)alloc((size_t)SEQ * KVD * 4);
  float* vf32 = (float*)alloc((size_t)SEQ * KVD * 4);
  __bf16* qhb = (__bf16*)alloc((size_t)NH * SEQ * HD * 2);
  __bf16* khb = (__bf16*)alloc((size_t)NKV * SEQ * HD * 2);
  __bf16* vtb = (__bf16*)alloc((size_t)NKV * HD * SEQ * 2);
  // Attention output (bf16, 16 MB) reuses the Q-f32 region (32 MB, dead
  // after rope_pack of Q). Stream-ordered, so no hazard.
  __bf16* obf = (__bf16*)qf32;

  f32_to_bf16<<<2048, 256, 0, stream>>>(x, xb, (size_t)SEQ * DM);
  f32_to_bf16<<<4096, 256, 0, stream>>>(wq, wqb, (size_t)DM * DM);
  f32_to_bf16<<<1024, 256, 0, stream>>>(wk, wkb, (size_t)KVD * DM);
  f32_to_bf16<<<1024, 256, 0, stream>>>(wv, wvb, (size_t)KVD * DM);
  f32_to_bf16<<<4096, 256, 0, stream>>>(wo, wob, (size_t)DM * DM);

  gemm_xwT<<<dim3(DM / 64, SEQ / 256), 256, 0, stream>>>(xb, wqb, qf32, SEQ, DM, DM);
  gemm_xwT<<<dim3(KVD / 64, SEQ / 256), 256, 0, stream>>>(xb, wkb, kf32, SEQ, KVD, DM);
  gemm_xwT<<<dim3(KVD / 64, SEQ / 256), 256, 0, stream>>>(xb, wvb, vf32, SEQ, KVD, DM);

  rope_pack<<<dim3(SEQ, NH), 64, 0, stream>>>(qf32, qhb, NH);
  rope_pack<<<dim3(SEQ, NKV), 64, 0, stream>>>(kf32, khb, NKV);
  pack_vt<<<dim3(SEQ, NKV), HD, 0, stream>>>(vf32, vtb);

  attn<<<dim3(SEQ / 128, NH), 256, 0, stream>>>(qhb, khb, vtb, obf);

  gemm_xwT<<<dim3(DM / 64, SEQ / 256), 256, 0, stream>>>(obf, wob, (float*)d_out,
                                                         SEQ, DM, DM);
}
